// Attention_17102559773274
// MI455X (gfx1250) — matrix operand order repaired and checked by reading, not compiled, not run on hardware
//
#include <hip/hip_runtime.h>

// ---------------------------------------------------------------------------
// Types for CDNA5 WMMA (wave32)
// ---------------------------------------------------------------------------
typedef __attribute__((ext_vector_type(16))) __bf16          v16bf;
typedef __attribute__((ext_vector_type(16))) unsigned short  v16u;
typedef __attribute__((ext_vector_type(8)))  unsigned short  v8u;
typedef __attribute__((ext_vector_type(4)))  unsigned short  v4u;
typedef __attribute__((ext_vector_type(8)))  float           v8f;
typedef __attribute__((ext_vector_type(4)))  float           v4f;

#define BDIM 2
#define SDIM 2048
#define HNUM 12
#define DDIM 64
#define NX   768
#define RPE  41   // 2*20+1

__device__ __forceinline__ unsigned short f2bf(float f) {
    return __builtin_bit_cast(unsigned short, (__bf16)f);
}
__device__ __forceinline__ float bf2f(unsigned short h) {
    return (float)__builtin_bit_cast(__bf16, h);
}
__device__ __forceinline__ v8f zero8() {
    v8f z = {0.f,0.f,0.f,0.f,0.f,0.f,0.f,0.f};
    return z;
}
__device__ __forceinline__ v8f wmma_bf16(v16u a, v16u b, v8f c) {
    return __builtin_amdgcn_wmma_f32_16x16x32_bf16(
        false, __builtin_bit_cast(v16bf, a),
        false, __builtin_bit_cast(v16bf, b),
        (short)0, c, false, false);
}

// 16 contiguous bf16 (two 16B vector loads)
__device__ __forceinline__ v16u frag_c(const unsigned short* p) {
    v8u lo = *(const v8u*)(p);
    v8u hi = *(const v8u*)(p + 8);
    return __builtin_shufflevector(lo, hi, 0,1,2,3,4,5,6,7,8,9,10,11,12,13,14,15);
}
// two separate 16B chunks (A/P fragment: K 16h+0..7 and K 16h+8..15 for lane half h)
__device__ __forceinline__ v16u frag_2(const unsigned short* p0, const unsigned short* p1) {
    v8u lo = *(const v8u*)(p0);
    v8u hi = *(const v8u*)(p1);
    return __builtin_shufflevector(lo, hi, 0,1,2,3,4,5,6,7,8,9,10,11,12,13,14,15);
}
// A-fragment (16x32): lane L row (L&15), 16 contiguous k at kb16
__device__ __forceinline__ v16u load_A_frag(const unsigned short* base, int ld, int lane) {
    const unsigned short* row = base + (lane & 15) * ld + ((lane >> 4) << 4);
    return frag_2(row, row + 8);
}

// ---------------------------------------------------------------------------
// Kernel 1: QKV projection.  C[4096,2304] = X[4096,768] @ W[768,2304] + bias
// Block tile 64(M) x 256(N), K step 32, 8 waves; each wave 32x64 (2x4 wmma).
// B staged transposed Bt[n][k] via 4x4 blocked transpose (b64 stores).
// ---------------------------------------------------------------------------
__global__ __launch_bounds__(256) void gemm_qkv(
    const float* __restrict__ X, const float* __restrict__ W,
    const float* __restrict__ bias,
    unsigned short* __restrict__ Q, unsigned short* __restrict__ K,
    unsigned short* __restrict__ V)
{
    __shared__ alignas(16) unsigned short As[64][40];    // [m][k]
    __shared__ alignas(16) unsigned short Bt[256][40];   // [n][k]
    const int t    = threadIdx.x;
    const int lane = t & 31;
    const int wave = t >> 5;
    const int wm   = wave >> 2;      // 0..1
    const int wn   = wave & 3;       // 0..3
    const int bm   = blockIdx.x * 64;
    const int bn   = blockIdx.y * 256;
    const int rot  = (lane >> 2) & 3;

    v8f acc[2][4];
#pragma unroll
    for (int i = 0; i < 2; ++i)
#pragma unroll
        for (int j = 0; j < 4; ++j) acc[i][j] = zero8();

    for (int k0 = 0; k0 < NX; k0 += 32) {
        {   // stage A: 64x32 fp32->bf16, 8 contiguous elems per thread
            int e = t * 8;
            int r = e >> 5, c = e & 31;
            v4f x0 = *(const v4f*)&X[(bm + r) * NX + k0 + c];
            v4f x1 = *(const v4f*)&X[(bm + r) * NX + k0 + c + 4];
            v8u pk;
#pragma unroll
            for (int j = 0; j < 4; ++j) { pk[j] = f2bf(x0[j]); pk[j + 4] = f2bf(x1[j]); }
            *(v8u*)&As[r][c] = pk;
        }
        // stage B transposed: 32x256, two 4x4 blocks per thread
#pragma unroll
        for (int p = 0; p < 2; ++p) {
            int pid = t + p * 256;           // 0..511
            int r0 = (pid >> 6) * 4;         // 0..28
            int c0 = (pid & 63) * 4;         // 0..252
            v4f w[4];
#pragma unroll
            for (int i = 0; i < 4; ++i)
                w[i] = *(const v4f*)&W[(k0 + r0 + i) * (3 * NX) + bn + c0];
#pragma unroll
            for (int jj = 0; jj < 4; ++jj) {
                int j = (jj + rot) & 3;      // rotate to spread LDS banks
                v4u pk;
#pragma unroll
                for (int i = 0; i < 4; ++i) pk[i] = f2bf(w[i][j]);
                *(v4u*)&Bt[c0 + j][r0] = pk;
            }
        }
        __syncthreads();

        const int kb16 = (lane >> 4) << 4;
        v16u af[2], bfr[4];
#pragma unroll
        for (int i = 0; i < 2; ++i) af[i]  = load_A_frag(&As[wm * 32 + i * 16][0], 40, lane);
#pragma unroll
        for (int j = 0; j < 4; ++j) bfr[j] = frag_c(&Bt[wn * 64 + j * 16 + (lane & 15)][kb16]);
#pragma unroll
        for (int i = 0; i < 2; ++i)
#pragma unroll
            for (int j = 0; j < 4; ++j)
                acc[i][j] = wmma_bf16(af[i], bfr[j], acc[i][j]);
        __syncthreads();
    }

    // epilogue: section / head / dest base are tile-uniform (256 | 768)
    const int sec  = bn / NX;
    unsigned short* __restrict__ dstbuf = (sec == 0) ? Q : (sec == 1) ? K : V;
    const int nxb   = bn - sec * NX;
    const int mbase = (lane >> 4) << 3;
    const int nloc  = lane & 15;
#pragma unroll
    for (int i = 0; i < 2; ++i) {
        int gm0 = bm + wm * 32 + i * 16 + mbase;       // rows gm0..gm0+7 (no b/s wrap)
        int b   = gm0 >> 11;
        int s0  = gm0 & (SDIM - 1);
#pragma unroll
        for (int j = 0; j < 4; ++j) {
            int nx = nxb + wn * 64 + j * 16;           // tile col base in section
            int h  = nx >> 6;                          // uniform per 16-wide tile
            int d0 = (nx & 63) + nloc;
            float bv = bias[bn + wn * 64 + j * 16 + nloc];
            unsigned short* dst = dstbuf + (((b * HNUM + h) * SDIM + s0) << 6) + d0;
#pragma unroll
            for (int r = 0; r < 8; ++r)
                dst[r << 6] = f2bf(acc[i][j][r] + bv);
        }
    }
}

// ---------------------------------------------------------------------------
// Kernel 2: RPE projection table  proj[b,s,h,r] = sum_d q[b,h,s,d]*rpek[r,d]
// ---------------------------------------------------------------------------
__global__ __launch_bounds__(256) void rpe_proj(
    const unsigned short* __restrict__ Q, const float* __restrict__ rpek,
    float* __restrict__ proj)
{
    __shared__ float rs[RPE * DDIM];
    __shared__ alignas(16) unsigned short qs[HNUM][DDIM];
    const int t = threadIdx.x;
    for (int i = t; i < RPE * DDIM; i += 256) rs[i] = rpek[i];
    const int bs = blockIdx.x;            // b*S + s
    const int b  = bs >> 11, s = bs & (SDIM - 1);
    if (t < HNUM * 8) {                   // 96 threads stage 12 q-rows
        int h = t >> 3, c = (t & 7) * 8;
        *(v8u*)&qs[h][c] =
            *(const v8u*)&Q[((size_t)((b * HNUM + h) * SDIM + s)) * DDIM + c];
    }
    __syncthreads();
    for (int o = t; o < HNUM * RPE; o += 256) {
        int h = o / RPE, r = o - h * RPE;
        float sum = 0.f;
#pragma unroll
        for (int d = 0; d < DDIM; ++d) sum += bf2f(qs[h][d]) * rs[r * DDIM + d];
        proj[((size_t)bs * HNUM + h) * RPE + r] = sum;
    }
}

// ---------------------------------------------------------------------------
// Kernel 3: fused flash attention with RPE bias gather.
// 4 waves/block, each wave owns 16 q rows; K step 32.
// K staged row-major [k][d]; V staged transposed [d][k] (4x4 blocks, b64);
// P published wave-locally via s_wait_dscnt (no block barrier).
// ---------------------------------------------------------------------------
__global__ __launch_bounds__(128) void flash_rpe(
    const unsigned short* __restrict__ Q, const unsigned short* __restrict__ Kb,
    const unsigned short* __restrict__ Vb, const float* __restrict__ proj,
    const int* __restrict__ dist, const float* __restrict__ mask,
    unsigned short* __restrict__ A)
{
    __shared__ alignas(16) unsigned short Ks[32][72];   // [k][d]
    __shared__ alignas(16) unsigned short Vt[64][40];   // [d][k]
    __shared__ alignas(16) unsigned short Ps[4][16][32];

    const int t     = threadIdx.x;
    const int lane  = t & 31;
    const int wave  = t >> 5;
    const int blk   = blockIdx.x;                // b*H*32 + h*32 + qt
    const int qt    = blk & 31;
    const int h     = (blk >> 5) % HNUM;
    const int b     = blk / (HNUM * 32);
    const int bh    = b * HNUM + h;
    const int q0    = qt * 64 + wave * 16;
    const int nloc  = lane & 15;
    const int mbase = (lane >> 4) << 3;
    const int kb8   = (lane >> 4) << 3;
    const int kb16  = (lane >> 4) << 4;
    const int rot   = (lane >> 2) & 3;

    // Q A-fragments (two 16x32 chunks over head dim)
    v16u qf[2];
    {
        const unsigned short* qrow = Q + ((size_t)(bh * SDIM) + q0 + nloc) * DDIM + kb16;
        qf[0] = frag_2(qrow,      qrow + 8);
        qf[1] = frag_2(qrow + 32, qrow + 40);
    }

    float mrow[8], lrow[8];
#pragma unroll
    for (int r = 0; r < 8; ++r) { mrow[r] = -1e30f; lrow[r] = 0.f; }
    v8f o[4] = { zero8(), zero8(), zero8(), zero8() };

    for (int kt = 0; kt < SDIM; kt += 32) {
        {   // stage K row-major (b128 copies)
            const unsigned short* kg = Kb + ((size_t)(bh * SDIM) + kt) * DDIM;
            int e = t * 16;
            int r = e >> 6, c0 = e & 63;
            *(v8u*)&Ks[r][c0]     = *(const v8u*)&kg[e];
            *(v8u*)&Ks[r][c0 + 8] = *(const v8u*)&kg[e + 8];
        }
        {   // stage V transposed: 4x4 block per thread, b64 stores
            const unsigned short* vg = Vb + ((size_t)(bh * SDIM) + kt) * DDIM;
            int r0 = (t >> 4) * 4;          // 0..28
            int c0 = (t & 15) * 4;          // 0..60
            v4u vv[4];
#pragma unroll
            for (int i = 0; i < 4; ++i)
                vv[i] = *(const v4u*)&vg[(r0 + i) * DDIM + c0];
#pragma unroll
            for (int jj = 0; jj < 4; ++jj) {
                int j = (jj + rot) & 3;
                v4u pk;
#pragma unroll
                for (int i = 0; i < 4; ++i) pk[i] = vv[i][j];
                *(v4u*)&Vt[c0 + j][r0] = pk;
            }
        }
        __syncthreads();

        // scores: two 16x16 tiles (k columns 0-15 and 16-31)
        v8f s[2];
#pragma unroll
        for (int nt = 0; nt < 2; ++nt) {
            v8f acc = zero8();
#pragma unroll
            for (int dh = 0; dh < 2; ++dh) {
                v16u bk = frag_c(&Ks[nt * 16 + nloc][dh * 32 + kb16]);
                acc = wmma_bf16(qf[dh], bk, acc);
            }
            s[nt] = acc;
        }

        // bias gather + mask + scale
        float sv[2][8];
#pragma unroll
        for (int nt = 0; nt < 2; ++nt) {
            int kg = kt + nt * 16 + nloc;
            float mk = mask[b * SDIM + kg];
#pragma unroll
            for (int r = 0; r < 8; ++r) {
                int qg  = q0 + mbase + r;
                int dix = dist[((size_t)(b * SDIM) + qg) * SDIM + kg] + 20;
                float bias = proj[(((size_t)(b * SDIM) + qg) * HNUM + h) * RPE + dix];
                sv[nt][r] = (s[nt][r] + bias) * 0.125f + mk;
            }
        }

        // online softmax (rows live across 16-lane halves)
#pragma unroll
        for (int r = 0; r < 8; ++r) {
            float mx = fmaxf(sv[0][r], sv[1][r]);
#pragma unroll
            for (int off = 8; off >= 1; off >>= 1)
                mx = fmaxf(mx, __shfl_xor(mx, off, 16));
            float mnew  = fmaxf(mrow[r], mx);
            float alpha = __expf(mrow[r] - mnew);
            mrow[r] = mnew;
            float p0 = __expf(sv[0][r] - mnew);
            float p1 = __expf(sv[1][r] - mnew);
            float ps = p0 + p1;
#pragma unroll
            for (int off = 8; off >= 1; off >>= 1)
                ps += __shfl_xor(ps, off, 16);
            lrow[r] = lrow[r] * alpha + ps;
#pragma unroll
            for (int dt = 0; dt < 4; ++dt) o[dt][r] *= alpha;
            Ps[wave][mbase + r][nloc]      = f2bf(p0);
            Ps[wave][mbase + r][16 + nloc] = f2bf(p1);
        }
        // Ps is wave-private: wave-local LDS drain instead of a block barrier
        asm volatile("s_wait_dscnt 0" ::: "memory");

        // P (16x32) @ V (32x64): 4 wmmas, V fragments contiguous from Vt
        v16u pf;
        {
            const unsigned short* pr = &Ps[wave][nloc][kb16];
            pf = frag_2(pr, pr + 8);
        }
#pragma unroll
        for (int dt = 0; dt < 4; ++dt) {
            v16u bv = frag_c(&Vt[dt * 16 + nloc][kb16]);
            o[dt] = wmma_bf16(pf, bv, o[dt]);
        }
        __syncthreads();   // Ks/Vt consumed before next staging
    }

    // normalize and store to A [B,S,H*D] (bf16)
#pragma unroll
    for (int r = 0; r < 8; ++r) {
        float inv = 1.0f / lrow[r];
        int qg = q0 + mbase + r;
        unsigned short* dst = A + ((size_t)(b * SDIM) + qg) * NX + h * DDIM;
#pragma unroll
        for (int dt = 0; dt < 4; ++dt)
            dst[dt * 16 + nloc] = f2bf(o[dt][r] * inv);
    }
}

// ---------------------------------------------------------------------------
// Kernel 4: output projection.  out[4096,768] = A(bf16) @ Wproj + bias (fp32)
// Block tile 64 x 256, 8 waves, each 32x64.
// ---------------------------------------------------------------------------
__global__ __launch_bounds__(256) void gemm_proj(
    const unsigned short* __restrict__ Ain, const float* __restrict__ W,
    const float* __restrict__ bias, float* __restrict__ out)
{
    __shared__ alignas(16) unsigned short As[64][40];
    __shared__ alignas(16) unsigned short Bt[256][40];
    const int t    = threadIdx.x;
    const int lane = t & 31;
    const int wave = t >> 5;
    const int wm   = wave >> 2;
    const int wn   = wave & 3;
    const int bm   = blockIdx.x * 64;
    const int bn   = blockIdx.y * 256;
    const int rot  = (lane >> 2) & 3;

    v8f acc[2][4];
#pragma unroll
    for (int i = 0; i < 2; ++i)
#pragma unroll
        for (int j = 0; j < 4; ++j) acc[i][j] = zero8();

    for (int k0 = 0; k0 < NX; k0 += 32) {
        {   // stage A (already bf16)
            int e = t * 8;
            int r = e >> 5, c = e & 31;
            *(v8u*)&As[r][c] = *(const v8u*)&Ain[(bm + r) * NX + k0 + c];
        }
#pragma unroll
        for (int p = 0; p < 2; ++p) {    // stage B transposed, fp32 -> bf16
            int pid = t + p * 256;
            int r0 = (pid >> 6) * 4;
            int c0 = (pid & 63) * 4;
            v4f w[4];
#pragma unroll
            for (int i = 0; i < 4; ++i)
                w[i] = *(const v4f*)&W[(k0 + r0 + i) * NX + bn + c0];
#pragma unroll
            for (int jj = 0; jj < 4; ++jj) {
                int j = (jj + rot) & 3;
                v4u pk;
#pragma unroll
                for (int i = 0; i < 4; ++i) pk[i] = f2bf(w[i][j]);
                *(v4u*)&Bt[c0 + j][r0] = pk;
            }
        }
        __syncthreads();

        const int kb16 = (lane >> 4) << 4;
        v16u af[2], bfr[4];
#pragma unroll
        for (int i = 0; i < 2; ++i) af[i]  = load_A_frag(&As[wm * 32 + i * 16][0], 40, lane);
#pragma unroll
        for (int j = 0; j < 4; ++j) bfr[j] = frag_c(&Bt[wn * 64 + j * 16 + (lane & 15)][kb16]);
#pragma unroll
        for (int i = 0; i < 2; ++i)
#pragma unroll
            for (int j = 0; j < 4; ++j)
                acc[i][j] = wmma_bf16(af[i], bfr[j], acc[i][j]);
        __syncthreads();
    }

    const int mbase = (lane >> 4) << 3;
    const int nloc  = lane & 15;
#pragma unroll
    for (int i = 0; i < 2; ++i) {
        int gm0 = bm + wm * 32 + i * 16 + mbase;
#pragma unroll
        for (int j = 0; j < 4; ++j) {
            int gn = bn + wn * 64 + j * 16 + nloc;
            float bv = bias[gn];
            float* dst = out + gm0 * NX + gn;
#pragma unroll
            for (int r = 0; r < 8; ++r)
                dst[r * NX] = acc[i][j][r] + bv;
        }
    }
}

// ---------------------------------------------------------------------------
// Launch
// ---------------------------------------------------------------------------
extern "C" void kernel_launch(void* const* d_in, const int* in_sizes, int n_in,
                              void* d_out, int out_size, void* d_ws, size_t ws_size,
                              hipStream_t stream) {
    const float* x     = (const float*)d_in[0];
    const float* amask = (const float*)d_in[1];
    const int*   pdist = (const int*)  d_in[2];
    const float* Wattn = (const float*)d_in[3];
    const float* battn = (const float*)d_in[4];
    const float* Wproj = (const float*)d_in[5];
    const float* bproj = (const float*)d_in[6];
    const float* rpek  = (const float*)d_in[7];
    float* out = (float*)d_out;

    const size_t NE = (size_t)BDIM * SDIM * NX;          // 3,145,728
    unsigned short* qb = (unsigned short*)d_ws;
    unsigned short* kb = qb + NE;
    unsigned short* vb = kb + NE;
    unsigned short* ab = vb + NE;
    float* proj = (float*)(ab + NE);                     // B*S*H*41 floats

    gemm_qkv<<<dim3(64, 9), 256, 0, stream>>>(x, Wattn, battn, qb, kb, vb);
    rpe_proj<<<BDIM * SDIM, 256, 0, stream>>>(qb, rpek, proj);
    flash_rpe<<<BDIM * HNUM * (SDIM / 64), 128, 0, stream>>>(qb, kb, vb, proj,
                                                             pdist, amask, ab);
    gemm_proj<<<dim3(64, 3), 256, 0, stream>>>(ab, Wproj, bproj, out);
}
